// Matching_67078799229472
// MI455X (gfx1250) — compile-verified, matrix operand order checked by reading, not run on hardware
//
#include <hip/hip_runtime.h>
#include <math.h>

typedef __attribute__((ext_vector_type(16))) _Float16 v16h;
typedef __attribute__((ext_vector_type(8)))  float    v8f;

#define BATCH 2
#define MROWS 1500
#define NCOLS 4096
#define KDIM  256
#define MP1   1501
#define NP1   4097
#define VSTRIDE 4104      // NP1 padded to multiple of 8 -> 16B-aligned float4 rows
#define THRV  0.1f

// workspace layout (floats)
#define U_OFF    0        // BATCH*MP1     = 3002
#define V_OFF    4096     // BATCH*VSTRIDE = 8208
#define RMAX_OFF 16384    // BATCH*MROWS   = 3000
#define CMAX_OFF 20480    // BATCH*NCOLS   = 8192
#define ZERO_CNT 16384

// ---------------------------------------------------------------- init u=v=0
__global__ __launch_bounds__(256) void zero_uv_kernel(float* __restrict__ ws) {
  int i = blockIdx.x * 256 + threadIdx.x;
  if (i < ZERO_CNT) ws[i] = 0.0f;
}

// ------------------------------------------------- f32 -> f16 WMMA fragment
// 16-bit operand layout (ISA 7.12.2): lane&15 = row, K split across lane
// halves; vector slot i -> K = (i>>3)*16 + ((lane>>4)<<3) + (i&7).
__device__ inline v16h load_frag16(const float* __restrict__ rowp, int kb0) {
  float4 c0 = *reinterpret_cast<const float4*>(rowp + kb0);
  float4 c1 = *reinterpret_cast<const float4*>(rowp + kb0 + 4);
  float4 c2 = *reinterpret_cast<const float4*>(rowp + kb0 + 16);
  float4 c3 = *reinterpret_cast<const float4*>(rowp + kb0 + 20);
  v16h f;
  f[0]=(_Float16)c0.x;  f[1]=(_Float16)c0.y;  f[2]=(_Float16)c0.z;  f[3]=(_Float16)c0.w;
  f[4]=(_Float16)c1.x;  f[5]=(_Float16)c1.y;  f[6]=(_Float16)c1.z;  f[7]=(_Float16)c1.w;
  f[8]=(_Float16)c2.x;  f[9]=(_Float16)c2.y;  f[10]=(_Float16)c2.z; f[11]=(_Float16)c2.w;
  f[12]=(_Float16)c3.x; f[13]=(_Float16)c3.y; f[14]=(_Float16)c3.z; f[15]=(_Float16)c3.w;
  return f;
}

// --------------------------------------------------------- sim = A * B^T * s
__global__ __launch_bounds__(128)
void gemm_sim_kernel(const float* __restrict__ f3d, const float* __restrict__ fq,
                     float* __restrict__ sim, float scale)
{
  const int lane = threadIdx.x & 31;
  const int wave = threadIdx.x >> 5;
  const int b    = blockIdx.z;
  const int m0   = (blockIdx.y * 4 + wave) * 16;
  const int n0   = blockIdx.x * 64;
  const int half = lane >> 4;
  const int r15  = lane & 15;

  const float* A  = f3d + (size_t)b * MROWS * KDIM;
  const float* Bq = fq  + (size_t)b * NCOLS * KDIM;

  v8f acc[4] = {};

  const int   arow = m0 + r15;
  const bool  aok  = arow < MROWS;
  const float* arp  = A  + (size_t)arow * KDIM;
  const float* brp0 = Bq + (size_t)(n0 + r15) * KDIM;

  for (int kk = 0; kk < KDIM; kk += 32) {
    const int kb0 = kk + half * 8;
    v16h afrag = {};
    if (aok) afrag = load_frag16(arp, kb0);
#pragma unroll
    for (int t = 0; t < 4; ++t) {
      v16h bfrag = load_frag16(brp0 + (size_t)t * 16 * KDIM, kb0);
      acc[t] = __builtin_amdgcn_wmma_f32_16x16x32_f16(
          false, afrag, false, bfrag, (short)0, acc[t], false, false);
    }
  }
#pragma unroll
  for (int t = 0; t < 4; ++t) {
#pragma unroll
    for (int r = 0; r < 8; ++r) {
      const int Mi = m0 + half * 8 + r;             // C layout: lanes16-31 -> M+8
      if (Mi < MROWS)
        sim[((size_t)b * MROWS + Mi) * NCOLS + n0 + t * 16 + r15] = acc[t][r] * scale;
    }
  }
}

// ------------------------------------------- Sinkhorn phase A: update u[b,l]
// One wave per row l in [0,1500]; float4 lanes; s==4096 (dustbin col) peeled
// to lane 0. l==1500 is the dustbin row (all alpha).
__global__ __launch_bounds__(128)
void sinkhorn_row_kernel(const float* __restrict__ sim, const float* __restrict__ alpha_p,
                         float* __restrict__ u, const float* __restrict__ v,
                         float norm, float logn_norm)
{
  const int lane = threadIdx.x & 31;
  const int wave = threadIdx.x >> 5;
  const int l = blockIdx.x * 4 + wave;
  if (l >= MP1) return;
  const int b = blockIdx.y;
  const float alpha = *alpha_p;
  const float* vb   = v + b * VSTRIDE;
  const float* srow = sim + ((size_t)b * MROWS + l) * NCOLS;
  const bool  isrow = (l < MROWS);
  const float xbin  = alpha + vb[NCOLS];            // s == 4096 term

  float m = (lane == 0) ? xbin : -3.0e38f;
  for (int s = lane * 4; s < NCOLS; s += 128) {
    float4 vv = *reinterpret_cast<const float4*>(vb + s);
    float4 ss;
    if (isrow) ss = *reinterpret_cast<const float4*>(srow + s);
    else       ss = make_float4(alpha, alpha, alpha, alpha);
    m = fmaxf(m, fmaxf(fmaxf(ss.x + vv.x, ss.y + vv.y),
                       fmaxf(ss.z + vv.z, ss.w + vv.w)));
  }
  for (int off = 16; off > 0; off >>= 1) m = fmaxf(m, __shfl_xor(m, off, 32));

  float t = (lane == 0) ? __expf(xbin - m) : 0.0f;
  for (int s = lane * 4; s < NCOLS; s += 128) {
    float4 vv = *reinterpret_cast<const float4*>(vb + s);
    float4 ss;
    if (isrow) ss = *reinterpret_cast<const float4*>(srow + s);
    else       ss = make_float4(alpha, alpha, alpha, alpha);
    t += __expf(ss.x + vv.x - m) + __expf(ss.y + vv.y - m)
       + __expf(ss.z + vv.z - m) + __expf(ss.w + vv.w - m);
  }
  for (int off = 16; off > 0; off >>= 1) t += __shfl_xor(t, off, 32);

  if (lane == 0) u[b * MP1 + l] = (isrow ? norm : logn_norm) - (m + __logf(t));
}

// ------------------------------------------- Sinkhorn phase B: update v[b,s]
// 64 columns x 16 row-chunks = 1024 threads/block (fills a WGP with 32 waves);
// coalesced column reads; LDS logsumexp merge across the 16 chunks.
__global__ __launch_bounds__(1024)
void sinkhorn_col_kernel(const float* __restrict__ sim, const float* __restrict__ alpha_p,
                         const float* __restrict__ u, float* __restrict__ v,
                         float norm, float logm_norm)
{
  __shared__ float sm[16][64];
  __shared__ float ssum[16][64];
  const int tx = threadIdx.x, ty = threadIdx.y;
  const int s = blockIdx.x * 64 + tx;
  const int b = blockIdx.y;
  const float alpha = *alpha_p;
  const float* ub = u + b * MP1;
  const bool sv    = (s < NP1);
  const bool inner = (s < NCOLS);
  const int l0 = ty * 94;                    // 16*94 = 1504 >= 1501
  const int l1 = min(MP1, l0 + 94);

  float m = -3.0e38f;
  float t = 0.0f;
  if (sv) {
    for (int l = l0; l < l1; ++l) {
      float base = (inner && l < MROWS) ? sim[((size_t)b * MROWS + l) * NCOLS + s] : alpha;
      m = fmaxf(m, base + ub[l]);
    }
    for (int l = l0; l < l1; ++l) {
      float base = (inner && l < MROWS) ? sim[((size_t)b * MROWS + l) * NCOLS + s] : alpha;
      t += __expf(base + ub[l] - m);
    }
  }
  sm[ty][tx] = m;  ssum[ty][tx] = t;
  __syncthreads();
  if (ty == 0 && sv) {
    float M = sm[0][tx];
#pragma unroll
    for (int j = 1; j < 16; ++j) M = fmaxf(M, sm[j][tx]);
    float S = 0.0f;
#pragma unroll
    for (int j = 0; j < 16; ++j) S += ssum[j][tx] * __expf(sm[j][tx] - M);
    v[b * VSTRIDE + s] = (inner ? norm : logm_norm) - (M + __logf(S));
  }
}

// --------------------------------- conf = sim + u + v - norm (in place, vec4)
__global__ __launch_bounds__(256)
void conf_finalize_kernel(float* __restrict__ conf, const float* __restrict__ u,
                          const float* __restrict__ v, float norm)
{
  const int i   = blockIdx.x * 256 + threadIdx.x;   // one float4 each; grid exact
  const int s4  = i & 1023;                          // 4096/4 columns
  const int row = i >> 10;                           // b*1500 + l
  const int b   = row / MROWS;
  const int l   = row - b * MROWS;
  const float add = u[b * MP1 + l] - norm;
  const float4 vv = *reinterpret_cast<const float4*>(v + b * VSTRIDE + s4 * 4);
  float4* p = reinterpret_cast<float4*>(conf) + i;
  float4 x = *p;
  x.x += add + vv.x;
  x.y += add + vv.y;
  x.z += add + vv.z;
  x.w += add + vv.w;
  *p = x;
}

// ------------------------------------------------------------- matching step
__global__ __launch_bounds__(128)
void rowmax_kernel(const float* __restrict__ conf, float* __restrict__ rmax)
{
  const int lane = threadIdx.x & 31, wave = threadIdx.x >> 5;
  const int l = blockIdx.x * 4 + wave;               // grid.x=375 -> l < 1500
  const int b = blockIdx.y;
  const float* crow = conf + ((size_t)b * MROWS + l) * NCOLS;
  float m = -3.0e38f;
  for (int s = lane * 4; s < NCOLS; s += 128) {
    float4 c = *reinterpret_cast<const float4*>(crow + s);
    m = fmaxf(m, fmaxf(fmaxf(c.x, c.y), fmaxf(c.z, c.w)));
  }
  for (int off = 16; off > 0; off >>= 1) m = fmaxf(m, __shfl_xor(m, off, 32));
  if (lane == 0) rmax[b * MROWS + l] = m;
}

__global__ __launch_bounds__(1024)
void colmax_kernel(const float* __restrict__ conf, float* __restrict__ cmax)
{
  __shared__ float sm[16][64];
  const int tx = threadIdx.x, ty = threadIdx.y;
  const int s = blockIdx.x * 64 + tx;                // grid.x=64 -> s < 4096
  const int b = blockIdx.y;
  float m = -3.0e38f;
  const int l0 = ty * 94;
  const int l1 = min(MROWS, l0 + 94);
  for (int l = l0; l < l1; ++l)
    m = fmaxf(m, conf[((size_t)b * MROWS + l) * NCOLS + s]);
  sm[ty][tx] = m;
  __syncthreads();
  if (ty == 0) {
#pragma unroll
    for (int j = 1; j < 16; ++j) m = fmaxf(m, sm[j][tx]);
    cmax[b * NCOLS + s] = m;
  }
}

__global__ __launch_bounds__(128)
void match_kernel(const float* __restrict__ conf, const float* __restrict__ rmax,
                  const float* __restrict__ cmax, float* __restrict__ mask_out,
                  float* __restrict__ jid_out, float* __restrict__ mconf_out)
{
  const int lane = threadIdx.x & 31, wave = threadIdx.x >> 5;
  const int l = blockIdx.x * 4 + wave;               // grid.x=375
  const int b = blockIdx.y;
  const float* crow = conf + ((size_t)b * MROWS + l) * NCOLS;
  const float rm = rmax[b * MROWS + l];
  const float* cm = cmax + b * NCOLS;
  int best = 0x7fffffff;                             // first True index (argmax of mask)
  for (int s = lane * 4; s < NCOLS; s += 128) {
    float4 c = *reinterpret_cast<const float4*>(crow + s);
    float4 g = *reinterpret_cast<const float4*>(cm + s);
    if (c.x > THRV && c.x == rm && c.x == g.x) best = min(best, s);
    if (c.y > THRV && c.y == rm && c.y == g.y) best = min(best, s + 1);
    if (c.z > THRV && c.z == rm && c.z == g.z) best = min(best, s + 2);
    if (c.w > THRV && c.w == rm && c.w == g.w) best = min(best, s + 3);
  }
  for (int off = 16; off > 0; off >>= 1) best = min(best, __shfl_xor(best, off, 32));
  if (lane == 0) {
    const int o = b * MROWS + l;
    if (best < NCOLS) {
      mask_out[o] = 1.0f; jid_out[o] = (float)best; mconf_out[o] = crow[best];
    } else {
      mask_out[o] = 0.0f; jid_out[o] = 0.0f;        mconf_out[o] = 0.0f;
    }
  }
}

// ---------------------------------------------------------------------------
extern "C" void kernel_launch(void* const* d_in, const int* in_sizes, int n_in,
                              void* d_out, int out_size, void* d_ws, size_t ws_size,
                              hipStream_t stream) {
  (void)in_sizes; (void)n_in; (void)out_size; (void)ws_size;
  const float* f3d   = (const float*)d_in[0];
  const float* fq    = (const float*)d_in[1];
  const float* alpha = (const float*)d_in[2];

  float* out       = (float*)d_out;
  float* conf      = out;                                   // 2*1500*4096
  float* mask_out  = out + (size_t)BATCH * MROWS * NCOLS;   // 3000
  float* jid_out   = mask_out + BATCH * MROWS;              // 3000
  float* mconf_out = jid_out  + BATCH * MROWS;              // 3000

  float* ws   = (float*)d_ws;
  float* u    = ws + U_OFF;
  float* v    = ws + V_OFF;
  float* rmax = ws + RMAX_OFF;
  float* cmax = ws + CMAX_OFF;

  const float norm      = -logf((float)(MROWS + NCOLS));
  const float logn_norm = logf((float)NCOLS) + norm;
  const float logm_norm = logf((float)MROWS) + norm;
  const float scale     = 1.0f / (256.0f * sqrtf(1500.0f)); // /sqrt(C) twice, /sqrt(L)

  zero_uv_kernel<<<ZERO_CNT / 256, 256, 0, stream>>>(ws);
  gemm_sim_kernel<<<dim3(NCOLS / 64, 24, BATCH), 128, 0, stream>>>(f3d, fq, conf, scale);
  for (int it = 0; it < 100; ++it) {
    sinkhorn_row_kernel<<<dim3(376, BATCH), 128, 0, stream>>>(conf, alpha, u, v, norm, logn_norm);
    sinkhorn_col_kernel<<<dim3(65, BATCH), dim3(64, 16), 0, stream>>>(conf, alpha, u, v, norm, logm_norm);
  }
  conf_finalize_kernel<<<12000, 256, 0, stream>>>(conf, u, v, norm);
  rowmax_kernel<<<dim3(375, BATCH), 128, 0, stream>>>(conf, rmax);
  colmax_kernel<<<dim3(64, BATCH), dim3(64, 16), 0, stream>>>(conf, cmax);
  match_kernel<<<dim3(375, BATCH), 128, 0, stream>>>(conf, rmax, cmax,
                                                     mask_out, jid_out, mconf_out);
}
